// MambaModel_24781961298455
// MI455X (gfx1250) — matrix operand order repaired and checked by reading, compiled, not run on hardware
//
#include <hip/hip_runtime.h>
#include <hip/hip_bf16.h>

// ---------------- CDNA5 WMMA types ----------------
typedef __attribute__((ext_vector_type(16))) _Float16 v16h;
typedef __attribute__((ext_vector_type(8)))  float    v8f;
typedef __attribute__((ext_vector_type(4)))  float    vf4;

#define B_SZ     2
#define SEQ      2048
#define DMODEL   512
#define DINNER   1024
#define DSTATE   16
#define DTRANK   32
#define NTOK     (B_SZ * SEQ)

// ---------------------------------------------------------------------
// CDNA5 async copy helpers (ASYNCcnt-tracked, memory -> LDS)
// VDST operand = LDS byte address (HW adds LDS_BASE); VADDR = 32-bit
// byte offset; SADDR = 64-bit base (GVS mode).
// ---------------------------------------------------------------------
__device__ __forceinline__ void async_ld_b128(unsigned lds_addr,
                                              const float* gbase,
                                              unsigned goff_bytes)
{
    asm volatile("global_load_async_to_lds_b128 %0, %1, %2"
                 :: "v"(lds_addr), "v"(goff_bytes), "s"(gbase)
                 : "memory");
}

__device__ __forceinline__ void wait_async0()
{
    asm volatile("s_wait_asynccnt 0" ::: "memory");
}

// =====================================================================
// WMMA GEMM:  C[M,N] = A[M,K] @ W[N,K]^T   (fp32 in, f16 WMMA, fp32 acc)
// grid = (N/32, M/32), block = 32 (one wave computes a 32x32 tile
// = 2x2 WMMA sub-tiles, 4 v_wmma per 32-wide K-step).
// Operands staged HBM -> LDS via global_load_async_to_lds_b128,
// double-buffered, synchronized with s_wait_asynccnt.
//
// A-fragment per ISA: lane L -> row m = L&15, K = h*8+{0..7}, 16+h*8+{0..7}
// B-fragment per ISA: lane L -> col n = L&15, K = h*8+{0..7}, 16+h*8+{0..7}  (h = L>>4)
// Since W is [N,K] row-major, B's column n is W's row n: contiguous.
// =====================================================================
template<int ADD>
__global__ void __launch_bounds__(32) gemm_wmma_kernel(
    const float* __restrict__ A,
    const float* __restrict__ W,
    float* __restrict__ C,
    int N, int K)
{
    __shared__ float Abuf[2][32][32];   // [buf][m in tile][k in tile]
    __shared__ float Bbuf[2][32][32];   // [buf][n in tile][k in tile]

    const int lane = threadIdx.x;
    const int lo   = lane & 15;
    const int h    = lane >> 4;   // 0/1: lane half
    const int rl   = lane >> 3;   // 0..3: row-within-group for async copy
    const int cl   = lane & 7;    // 0..7: 16B column for async copy
    const int m0   = blockIdx.y * 32;
    const int n0   = blockIdx.x * 32;

    // LDS byte addresses (low 32 bits of generic pointer == LDS offset)
    const unsigned ldsA0 = (unsigned)(unsigned long long)(void*)&Abuf[0][0][0];
    const unsigned ldsB0 = (unsigned)(unsigned long long)(void*)&Bbuf[0][0][0];

    // Issue async copy of one 32x32 fp32 tile pair (A and W) into buf.
    // Tile = 4KB = 8 issues of b128 x 32 lanes.
    auto issue_tiles = [&](int buf, int k0) {
#pragma unroll
        for (int i = 0; i < 8; ++i) {
            const unsigned la = ldsA0 + buf * 4096 + i * 512 + lane * 16;
            const unsigned ga = (unsigned)(((m0 + i * 4 + rl) * K + k0) * 4) + cl * 16;
            async_ld_b128(la, A, ga);
        }
#pragma unroll
        for (int i = 0; i < 8; ++i) {
            const unsigned lb = ldsB0 + buf * 4096 + i * 512 + lane * 16;
            const unsigned gb = (unsigned)(((n0 + i * 4 + rl) * K + k0) * 4) + cl * 16;
            async_ld_b128(lb, W, gb);
        }
    };

    issue_tiles(0, 0);

    v8f acc[2][2] = {};
    int buf = 0;

    for (int k = 0; k < K; k += 32) {
        wait_async0();   // tile pair for this K-step resident in LDS

        // ---- fragments from LDS (ds_load_b128) with f32->f16 convert
        v16h af[2], bf[2];
#pragma unroll
        for (int mi = 0; mi < 2; ++mi) {
            const float* ar = &Abuf[buf][mi * 16 + lo][0];
            vf4 a0 = *(const vf4*)(ar + h * 8);
            vf4 a1 = *(const vf4*)(ar + h * 8 + 4);
            vf4 a2 = *(const vf4*)(ar + 16 + h * 8);
            vf4 a3 = *(const vf4*)(ar + 16 + h * 8 + 4);
#pragma unroll
            for (int i = 0; i < 4; ++i) {
                af[mi][i]      = (_Float16)a0[i];
                af[mi][4 + i]  = (_Float16)a1[i];
                af[mi][8 + i]  = (_Float16)a2[i];
                af[mi][12 + i] = (_Float16)a3[i];
            }
        }
#pragma unroll
        for (int ni = 0; ni < 2; ++ni) {
            const float* br = &Bbuf[buf][ni * 16 + lo][0];
            vf4 b0 = *(const vf4*)(br + h * 8);
            vf4 b1 = *(const vf4*)(br + h * 8 + 4);
            vf4 b2 = *(const vf4*)(br + 16 + h * 8);
            vf4 b3 = *(const vf4*)(br + 16 + h * 8 + 4);
#pragma unroll
            for (int i = 0; i < 4; ++i) {
                bf[ni][i]      = (_Float16)b0[i];
                bf[ni][4 + i]  = (_Float16)b1[i];
                bf[ni][8 + i]  = (_Float16)b2[i];
                bf[ni][12 + i] = (_Float16)b3[i];
            }
        }

        // prefetch next K-tile into the other buffer while WMMAs run
        if (k + 32 < K) issue_tiles(buf ^ 1, k + 32);

#pragma unroll
        for (int mi = 0; mi < 2; ++mi)
#pragma unroll
            for (int ni = 0; ni < 2; ++ni)
                acc[mi][ni] = __builtin_amdgcn_wmma_f32_16x16x32_f16(
                    false, af[mi], false, bf[ni],
                    (short)0, acc[mi][ni], false, false);

        buf ^= 1;
    }

    // ---- Store D: VGPR r, lane L -> M = r + h*8 (+16*mi), N = lo (+16*ni)
#pragma unroll
    for (int mi = 0; mi < 2; ++mi)
#pragma unroll
        for (int ni = 0; ni < 2; ++ni)
#pragma unroll
            for (int r = 0; r < 8; ++r) {
                const size_t idx =
                    (size_t)(m0 + mi * 16 + r + h * 8) * N + (n0 + ni * 16 + lo);
                float v = acc[mi][ni][r];
                if (ADD) v += C[idx];
                C[idx] = v;
            }
}

// =====================================================================
// input (B, DMODEL, S) -> x_state (B, S, DMODEL)
// =====================================================================
__global__ void transpose_in_kernel(const float* __restrict__ in,
                                    float* __restrict__ x)
{
    const int idx = blockIdx.x * blockDim.x + threadIdx.x; // over B*S*DMODEL
    const int d = idx % DMODEL;
    const int t = idx / DMODEL;
    const int s = t % SEQ;
    const int b = t / SEQ;
    x[idx] = in[((size_t)b * DMODEL + d) * SEQ + s];
}

// =====================================================================
// RMSNorm: one 256-thread block per token (D = DMODEL)
// =====================================================================
__global__ void rmsnorm_kernel(const float* __restrict__ x,
                               const float* __restrict__ w,
                               float* __restrict__ out)
{
    __shared__ float red[256];
    const int t = blockIdx.x;
    const float* row = x + (size_t)t * DMODEL;
    float s = 0.f;
    for (int i = threadIdx.x; i < DMODEL; i += 256) {
        float v = row[i];
        s += v * v;
    }
    red[threadIdx.x] = s;
    __syncthreads();
    for (int off = 128; off > 0; off >>= 1) {
        if (threadIdx.x < off) red[threadIdx.x] += red[threadIdx.x + off];
        __syncthreads();
    }
    const float rs = rsqrtf(red[0] / (float)DMODEL + 1e-5f);
    for (int i = threadIdx.x; i < DMODEL; i += 256)
        out[(size_t)t * DMODEL + i] = row[i] * rs * w[i];
}

// =====================================================================
// Causal depthwise conv1d (width 4) + SiLU on u half of xz.
// xz layout: [NTOK, 2*DINNER]; u = xz[:, :DINNER]
// =====================================================================
__global__ void conv_silu_kernel(const float* __restrict__ xz,
                                 const float* __restrict__ cw,
                                 const float* __restrict__ cb,
                                 float* __restrict__ u)
{
    const int idx = blockIdx.x * blockDim.x + threadIdx.x; // over NTOK*DINNER
    const int d = idx & (DINNER - 1);
    const int t = idx >> 10;
    const int s = t & (SEQ - 1);
    float acc = cb[d];
#pragma unroll
    for (int j = 0; j < 4; ++j) {
        const int sp = s - 3 + j;
        if (sp >= 0)
            acc += cw[d * 4 + j] * xz[(size_t)(t - 3 + j) * (2 * DINNER) + d];
    }
    const float sig = 1.f / (1.f + __expf(-acc));
    u[idx] = acc * sig;
}

// =====================================================================
// dt = softplus(x_dbl[:, :32] @ dtproj^T + dtproj_b)   (K=32, scalar fp32)
// =====================================================================
__global__ void dt_kernel(const float* __restrict__ xdbl,
                          const float* __restrict__ dtw,
                          const float* __restrict__ dtb,
                          float* __restrict__ dt)
{
    const int idx = blockIdx.x * blockDim.x + threadIdx.x; // over NTOK*DINNER
    const int d = idx & (DINNER - 1);
    const int t = idx >> 10;
    const float* xr = xdbl + (size_t)t * (DTRANK + 2 * DSTATE);
    const float* wr = dtw + d * DTRANK;
    float acc = dtb[d];
#pragma unroll
    for (int r = 0; r < DTRANK; ++r) acc += xr[r] * wr[r];
    dt[idx] = (acc > 20.f) ? acc : log1pf(__expf(acc));
}

// =====================================================================
// Selective scan fused with output gating: one thread per (b, d) channel,
// sequential over S; h[16] in registers. gate = (y + u*D) * silu(z).
// =====================================================================
__global__ void scan_gate_kernel(const float* __restrict__ dt,
                                 const float* __restrict__ u,
                                 const float* __restrict__ xdbl,
                                 const float* __restrict__ xz,
                                 const float* __restrict__ A_log,
                                 const float* __restrict__ Dp,
                                 float* __restrict__ gate)
{
    const int idx = blockIdx.x * blockDim.x + threadIdx.x; // over B*DINNER
    const int d = idx & (DINNER - 1);
    const int b = idx >> 10;

    float a[DSTATE];
#pragma unroll
    for (int i = 0; i < DSTATE; ++i)
        a[i] = -__expf(A_log[(size_t)d * DSTATE + i]);
    const float Dd = Dp[d];

    float h[DSTATE];
#pragma unroll
    for (int i = 0; i < DSTATE; ++i) h[i] = 0.f;

    for (int s = 0; s < SEQ; ++s) {
        const size_t t = (size_t)b * SEQ + s;
        const float dtv = dt[t * DINNER + d];
        const float uv  = u[t * DINNER + d];
        const float* xd = xdbl + t * (DTRANK + 2 * DSTATE);
        if (s + 1 < SEQ) {
            __builtin_prefetch(dt + (t + 1) * DINNER + d, 0, 1);
            __builtin_prefetch(u + (t + 1) * DINNER + d, 0, 1);
        }
        const float du = dtv * uv;
        float y = 0.f;
#pragma unroll
        for (int i = 0; i < DSTATE; ++i) {
            const float dA = __expf(dtv * a[i]);
            h[i] = dA * h[i] + du * xd[DTRANK + i];
            y += h[i] * xd[DTRANK + DSTATE + i];
        }
        y += uv * Dd;
        const float z = xz[t * (2 * DINNER) + DINNER + d];
        const float zsig = 1.f / (1.f + __expf(-z));
        gate[t * DINNER + d] = y * (z * zsig);
    }
}

// =====================================================================
// lengths = sum(mask); emb = tokens[b, lengths-1, :]
// =====================================================================
__global__ void emb_kernel(const unsigned char* __restrict__ mask,
                           const float* __restrict__ tokens,
                           float* __restrict__ emb)
{
    __shared__ int red[256];
    const int b = blockIdx.x;
    int cnt = 0;
    for (int i = threadIdx.x; i < SEQ; i += 256)
        cnt += mask[b * SEQ + i] ? 1 : 0;
    red[threadIdx.x] = cnt;
    __syncthreads();
    for (int off = 128; off > 0; off >>= 1) {
        if (threadIdx.x < off) red[threadIdx.x] += red[threadIdx.x + off];
        __syncthreads();
    }
    int idx = red[0] - 1;
    if (idx < 0) idx = 0;
    for (int i = threadIdx.x; i < DMODEL; i += 256)
        emb[b * DMODEL + i] = tokens[((size_t)b * SEQ + idx) * DMODEL + i];
}

// =====================================================================
// Host-side orchestration
// =====================================================================
extern "C" void kernel_launch(void* const* d_in, const int* in_sizes, int n_in,
                              void* d_out, int out_size, void* d_ws, size_t ws_size,
                              hipStream_t stream)
{
    const float*         input   = (const float*)d_in[0];
    const unsigned char* mask    = (const unsigned char*)d_in[1];
    const float*         in_proj = (const float*)d_in[2];
    const float*         conv_w  = (const float*)d_in[3];
    const float*         conv_b  = (const float*)d_in[4];
    const float*         xproj   = (const float*)d_in[5];
    const float*         dtw     = (const float*)d_in[6];
    const float*         dtb     = (const float*)d_in[7];
    const float*         A_log   = (const float*)d_in[8];
    const float*         Dp      = (const float*)d_in[9];
    const float*         outproj = (const float*)d_in[10];
    const float*         norm_w  = (const float*)d_in[11];
    const float*         fnorm_w = (const float*)d_in[12];
    float* out = (float*)d_out;

    const int XD = DTRANK + 2 * DSTATE; // 64

    float* ws      = (float*)d_ws;
    float* x_state = ws;                                   // NTOK*DMODEL
    float* xn      = x_state + (size_t)NTOK * DMODEL;      // NTOK*DMODEL
    float* xz      = xn      + (size_t)NTOK * DMODEL;      // NTOK*2*DINNER
    float* u       = xz      + (size_t)NTOK * 2 * DINNER;  // NTOK*DINNER
    float* xdbl    = u       + (size_t)NTOK * DINNER;      // NTOK*64
    float* dt      = xdbl    + (size_t)NTOK * XD;          // NTOK*DINNER
    float* gate    = dt      + (size_t)NTOK * DINNER;      // NTOK*DINNER

    transpose_in_kernel<<<(NTOK * DMODEL) / 256, 256, 0, stream>>>(input, x_state);

    for (int l = 0; l < 2; ++l) {
        rmsnorm_kernel<<<NTOK, 256, 0, stream>>>(x_state, norm_w + l * DMODEL, xn);

        // xz = xn @ in_proj[l]^T : [NTOK, 2*DINNER]
        gemm_wmma_kernel<0><<<dim3((2 * DINNER) / 32, NTOK / 32), 32, 0, stream>>>(
            xn, in_proj + (size_t)l * 2 * DINNER * DMODEL, xz,
            2 * DINNER, DMODEL);

        conv_silu_kernel<<<(NTOK * DINNER) / 256, 256, 0, stream>>>(
            xz, conv_w + l * DINNER * 4, conv_b + l * DINNER, u);

        // x_dbl = u @ xproj[l]^T : [NTOK, 64]
        gemm_wmma_kernel<0><<<dim3(XD / 32, NTOK / 32), 32, 0, stream>>>(
            u, xproj + (size_t)l * XD * DINNER, xdbl, XD, DINNER);

        dt_kernel<<<(NTOK * DINNER) / 256, 256, 0, stream>>>(
            xdbl, dtw + l * DINNER * DTRANK, dtb + l * DINNER, dt);

        scan_gate_kernel<<<(B_SZ * DINNER) / 256, 256, 0, stream>>>(
            dt, u, xdbl, xz, A_log + (size_t)l * DINNER * DSTATE, Dp + l * DINNER, gate);

        // x_state += gate @ outproj[l]^T : [NTOK, DMODEL]
        gemm_wmma_kernel<1><<<dim3(DMODEL / 32, NTOK / 32), 32, 0, stream>>>(
            gate, outproj + (size_t)l * DMODEL * DINNER, x_state,
            DMODEL, DINNER);
    }

    rmsnorm_kernel<<<NTOK, 256, 0, stream>>>(x_state, fnorm_w, out);
    emb_kernel<<<B_SZ, 256, 0, stream>>>(mask, out, out + (size_t)NTOK * DMODEL);
}
